// AnswerEmbedding_60748017434746
// MI455X (gfx1250) — matrix operand order repaired and checked by reading, not compiled, not run on hardware
//
#include <hip/hip_runtime.h>
#include <hip/hip_bf16.h>
#include <math.h>

// ---------------------------------------------------------------------------
// CDNA5 (gfx1250) conv-seq2seq decoder forward pass.
//  - All large GEMMs on v_wmma_f32_16x16x32_f16 (f32 accumulate).
//  - Weights / encoder tensors pre-converted to f16 each call.
//  - Conv weight tiles fetched with the Tensor Data Mover (tensor_load_to_lds,
//    TENSORcnt); generic GEMM B tiles use GLOBAL_LOAD_ASYNC_TO_LDS_B128
//    (ASYNCcnt). A (activations) stay f32; staged via float4 -> f16 path.
// ---------------------------------------------------------------------------

typedef _Float16 v16h __attribute__((ext_vector_type(16)));
typedef _Float16 v8h  __attribute__((ext_vector_type(8)));
typedef _Float16 v4h  __attribute__((ext_vector_type(4)));
typedef float    v8f  __attribute__((ext_vector_type(8)));
typedef unsigned int v4u __attribute__((ext_vector_type(4)));
typedef int      v8i  __attribute__((ext_vector_type(8)));
typedef int      v4i  __attribute__((ext_vector_type(4)));

union F16Frag { v16h v; v8h h[2]; };

#define SCALE_F 0.5477225575051661f   // sqrt(0.3)

// Problem dims
static constexpr int Bsz  = 64;
static constexpr int SRC  = 100;
static constexpr int EMB  = 512;
static constexpr int HID  = 1024;
static constexpr int TRG  = 100;
static constexpr int OUTN = 10000;
static constexpr int NLAY = 8;
static constexpr int TPAD = 102;      // TRG + 2 zero pad rows for K=3 conv

// ---- workspace layout -----------------------------------------------------
static constexpr size_t OFF_M    = 0;                       //   6400
static constexpr size_t OFF_HT   = 8192;                    //   6400
static constexpr size_t OFF_TOK  = 16384;                   //  32768
static constexpr size_t OFF_EMB  = 65536;                   // 3,276,800
static constexpr size_t OFF_XPAD = OFF_EMB  + 3276800;      // 6,684,672
static constexpr size_t OFF_YGLU = OFF_XPAD + 6684672;      // 6,553,600
static constexpr size_t OFF_TEMB = OFF_YGLU + 6553600;      // 3,276,800
static constexpr size_t OFF_ATTN = OFF_TEMB + 3276800;      //   640,000
static constexpr size_t HALF_BASE = 20971520;               // f16 region (float offset)
static constexpr size_t H_WCONV = 0;                        // 50,331,648  [l*3+tap][2048][1024]
static constexpr size_t H_E2H   = H_WCONV + 50331648;
static constexpr size_t H_H2E   = H_E2H   + 524288;
static constexpr size_t H_AH2E  = H_H2E   + 524288;
static constexpr size_t H_AE2H  = H_AH2E  + 524288;
static constexpr size_t H_FC    = H_AE2H  + 524288;
static constexpr size_t H_ENCC  = H_FC    + 5120000;
static constexpr size_t H_ENCCB = H_ENCC  + 3276800;

// ---------------------------------------------------------------------------
__device__ __forceinline__ size_t row_off(int m, int ld, int mode) {
  if (mode) {                       // padded conv-activation buffer [B][102][HID]
    int b = m / TRG;
    int t = m - b * TRG;
    return (size_t)(b * TPAD + t + 1) * (size_t)ld;
  }
  return (size_t)m * (size_t)ld;
}

__device__ __forceinline__ unsigned lds_addr(const void* p) {
  // generic shared pointer: addr[31:0] is the LDS byte offset
  return (unsigned)(size_t)p;
}

__device__ __forceinline__ void async_ld_b128(unsigned lds, const void* gaddr) {
  asm volatile("global_load_async_to_lds_b128 %0, %1, off"
               :: "v"(lds), "v"(gaddr) : "memory");
}
__device__ __forceinline__ void wait_async0() {
  asm volatile("s_wait_asynccnt 0x0" ::: "memory");
}

// TDM: load a 2-D f16 tile [tile1 rows x tile0 cols], row stride `stride0`
// (elements), from global `gptr` into LDS byte address `ldsByte` (packed
// row-major, tile0 elements per row). D# per CDNA5 ISA ch.8; groups 2/3 zero.
// This toolchain exposes the 6-arg builtin:
//   (uint32x4 g0, int32x8 g1, int32x4, int32x4, int32x8, i32 cpol)
__device__ __forceinline__ void tdm_load_tile_f16(unsigned ldsByte, const void* gptr,
                                                  unsigned tile0, unsigned tile1,
                                                  unsigned stride0) {
  unsigned long long ga = (unsigned long long)gptr;
  const unsigned D0 = 1u << 20, D1 = 1u << 20;   // tensor dims (no clipping)
  v4u g0;
  g0[0] = 1u;                                                   // count=1
  g0[1] = ldsByte;                                              // lds_addr
  g0[2] = (unsigned)ga;                                         // global_addr[31:0]
  g0[3] = (unsigned)((ga >> 32) & 0x1FFFFFFull) | (2u << 30);   // addr[56:32] | type=2
  v8i g1;
  g1[0] = (int)(1u << 16);                                      // data_size = 2B
  g1[1] = (int)((D0 & 0xFFFFu) << 16);                          // tensor_dim0 lo
  g1[2] = (int)((D0 >> 16) | ((D1 & 0xFFFFu) << 16));           // dim0 hi | dim1 lo
  g1[3] = (int)((D1 >> 16) | (tile0 << 16));                    // dim1 hi | tile_dim0
  g1[4] = (int)(tile1 & 0xFFFFu);                               // tile_dim1 (tile_dim2=0)
  g1[5] = (int)stride0;                                         // tensor_dim0_stride lo
  g1[6] = 0;
  g1[7] = 0;
  v4i z4 = { 0, 0, 0, 0 };
  v8i z8 = { 0, 0, 0, 0, 0, 0, 0, 0 };
  __builtin_amdgcn_tensor_load_to_lds(g0, g1, z4, z4, z8, 0);
}

// ---------------------------------------------------------------------------
// Small kernels
// ---------------------------------------------------------------------------
__global__ void zero_fill_kernel(float* p, size_t n) {
  size_t i = (size_t)blockIdx.x * blockDim.x + threadIdx.x;
  size_t stride = (size_t)gridDim.x * blockDim.x;
  for (; i < n; i += stride) p[i] = 0.0f;
}

__global__ void cvt_f16_kernel(const float* __restrict__ in, _Float16* __restrict__ out, size_t n) {
  size_t i = (size_t)blockIdx.x * blockDim.x + threadIdx.x;
  size_t stride = (size_t)gridDim.x * blockDim.x;
  for (; i < n; i += stride) out[i] = (_Float16)in[i];
}

// conv_W [l][o][i][k] f32  ->  [l*3+k][o][i] f16 (contiguous along i)
__global__ void cvt_convw_kernel(const float* __restrict__ in, _Float16* __restrict__ out) {
  size_t total = (size_t)NLAY * 3 * 2048 * 1024;
  size_t i = (size_t)blockIdx.x * blockDim.x + threadIdx.x;
  size_t stride = (size_t)gridDim.x * blockDim.x;
  for (; i < total; i += stride) {
    size_t t  = i;
    int ci = (int)(t & 1023);  t >>= 10;
    int o  = (int)(t & 2047);  t >>= 11;
    int u  = (int)t;            // u = l*3 + k
    int k  = u % 3, l = u / 3;
    out[i] = (_Float16)in[(((size_t)l * 2048 + o) * 1024 + ci) * 3 + k];
  }
}

// m[b,s] = mean over e of enc_conved[b,s,e]   (one wave per row)
__global__ void mean512_kernel(const float* __restrict__ enc, float* __restrict__ out) {
  int row  = blockIdx.x * 4 + (threadIdx.x >> 5);
  int lane = threadIdx.x & 31;
  const float* p = enc + (size_t)row * EMB;
  float s = 0.f;
  for (int e = lane; e < EMB; e += 32) s += p[e];
  for (int off = 16; off > 0; off >>= 1) s += __shfl_down(s, off, 32);
  if (lane == 0) out[row] = s * (1.0f / (float)EMB);
}

// tiny GEMMs: C[M,N] = A[M,K] @ W[N,K]^T + bias
__global__ void small_gemm_kernel(const float* __restrict__ A, const float* __restrict__ W,
                                  const float* __restrict__ bias, float* __restrict__ C,
                                  int M, int N, int K) {
  int idx = blockIdx.x * blockDim.x + threadIdx.x;
  if (idx >= M * N) return;
  int m = idx / N, n = idx - m * N;
  float s = bias[n];
  const float* a = A + (size_t)m * K;
  const float* w = W + (size_t)n * K;
  for (int k = 0; k < K; k++) s += a[k] * w[k];
  C[idx] = s;
}

__global__ void embed_add_kernel(const float* __restrict__ tok, const float* __restrict__ pos,
                                 float* __restrict__ emb) {
  int idx = blockIdx.x * blockDim.x + threadIdx.x;
  if (idx >= Bsz * TRG * EMB) return;
  int e  = idx & (EMB - 1);
  int bt = idx >> 9;
  int t  = bt % TRG;
  int b  = bt / TRG;
  emb[idx] = tok[b * EMB + e] + pos[t * EMB + e];
}

// row-wise softmax over 100 elements (one wave per row)
__global__ void softmax100_kernel(float* __restrict__ attn) {
  int row  = blockIdx.x * 4 + (threadIdx.x >> 5);
  int lane = threadIdx.x & 31;
  float* p = attn + (size_t)row * SRC;
  float v[4];
  float mx = -1e30f;
  for (int i = 0; i < 4; i++) {
    int s = lane + i * 32;
    v[i] = (s < SRC) ? p[s] : -1e30f;
    mx = fmaxf(mx, v[i]);
  }
  for (int off = 16; off > 0; off >>= 1) mx = fmaxf(mx, __shfl_down(mx, off, 32));
  mx = __shfl(mx, 0, 32);
  float sum = 0.f;
  for (int i = 0; i < 4; i++) {
    int s = lane + i * 32;
    if (s < SRC) { v[i] = __expf(v[i] - mx); sum += v[i]; }
  }
  for (int off = 16; off > 0; off >>= 1) sum += __shfl_down(sum, off, 32);
  sum = __shfl(sum, 0, 32);
  float inv = 1.0f / sum;
  for (int i = 0; i < 4; i++) {
    int s = lane + i * 32;
    if (s < SRC) p[s] = v[i] * inv;
  }
}

// ---------------------------------------------------------------------------
// Generic WMMA GEMM:  C[M,N] = epi( A[M,K](f32) @ Wh^T(f16) + bias )
//   Wh element (n,k) at  Wh + n*nStride + k*kStride.
//   epi 0: none; 1: c=(c+aux1)*S; 2: c=((c+aux1)*S + Cold)*S
// Block 256 thr (8 waves), tile 128x128, K-step 32, wave = 2x4 wmma tiles.
// ---------------------------------------------------------------------------
struct GemmParams {
  const float*    A;  size_t aBatch; int lda; int aMode;
  const _Float16* Wh; size_t wBatch; size_t nStride; size_t kStride;
  const float*    bias;
  float*          C;  size_t cBatch; int ldc; int cMode;
  const float*    aux1;
  int M, N, K;
  int epi;
};

__global__ __launch_bounds__(256) void wmma_gemm_kernel(GemmParams p) {
  __shared__ __align__(16) _Float16 smA[128 * 32];
  __shared__ __align__(16) _Float16 smB[128 * 32];

  const int bat = blockIdx.z;
  const float*    A  = p.A  + (size_t)bat * p.aBatch;
  const _Float16* Wh = p.Wh + (size_t)bat * p.wBatch;
  float*          C  = p.C  + (size_t)bat * p.cBatch;

  const int m0 = blockIdx.y * 128;
  const int n0 = blockIdx.x * 128;
  const int tid  = threadIdx.x;
  const int wave = tid >> 5;
  const int lane = tid & 31;
  const int wm = wave & 3;        // 4 waves over M
  const int wn = wave >> 2;       // 2 waves over N
  const int r    = lane & 15;
  const int half = lane >> 4;

  const bool bAsync = (p.kStride == 1) && (n0 + 128 <= p.N);

  v8f acc[2][4] = {};

  for (int kk = 0; kk < p.K; kk += 32) {
    const bool fullK = (kk + 32 <= p.K);
    // ---- stage A tile 128x32 (f32 -> f16) ----
    if (fullK) {
#pragma unroll
      for (int idx = tid; idx < 128 * 8; idx += 256) {
        int row = idx >> 3, c4 = (idx & 7) << 2;
        int mg = m0 + row;
        float4 vv = make_float4(0.f, 0.f, 0.f, 0.f);
        if (mg < p.M) vv = *(const float4*)(A + row_off(mg, p.lda, p.aMode) + kk + c4);
        v4h h = { (_Float16)vv.x, (_Float16)vv.y, (_Float16)vv.z, (_Float16)vv.w };
        *(v4h*)&smA[row * 32 + c4] = h;
      }
    } else {
      for (int idx = tid; idx < 128 * 32; idx += 256) {
        int row = idx >> 5, col = idx & 31;
        int mg = m0 + row, kg = kk + col;
        float v = 0.f;
        if (mg < p.M && kg < p.K) v = A[row_off(mg, p.lda, p.aMode) + kg];
        smA[idx] = (_Float16)v;
      }
    }
    // ---- stage B tile 128x32 (f16) ----
    if (bAsync && fullK) {
      // 128 rows x 4 x 16B segments = 512 segs; 2 per thread, DMA to LDS
      int row = tid >> 2, seg = tid & 3;
      const _Float16* src = Wh + (size_t)(n0 + row) * p.nStride + kk + seg * 8;
      unsigned dst = lds_addr(&smB[row * 32 + seg * 8]);
      async_ld_b128(dst, src);
      async_ld_b128(dst + 64 * 64, src + (size_t)64 * p.nStride);
      wait_async0();
    } else {
      for (int idx = tid; idx < 128 * 32; idx += 256) {
        int row = idx >> 5, col = idx & 31;
        int ng = n0 + row, kg = kk + col;
        _Float16 v = (_Float16)0.f;
        if (ng < p.N && kg < p.K) v = Wh[(size_t)ng * p.nStride + (size_t)kg * p.kStride];
        smB[idx] = v;
      }
    }
    __syncthreads();

    F16Frag af[2], bf[4];
    for (int i = 0; i < 2; i++) {
      int row = wm * 32 + i * 16 + r;
      af[i].h[0] = *(const v8h*)&smA[row * 32 + half * 8];        // K = half*8 .. +7
      af[i].h[1] = *(const v8h*)&smA[row * 32 + 16 + half * 8];   // K = 16+half*8 .. +7
    }
    for (int j = 0; j < 4; j++) {
      int row = wn * 64 + j * 16 + r;
      bf[j].h[0] = *(const v8h*)&smB[row * 32 + half * 8];        // K = half*8 .. +7
      bf[j].h[1] = *(const v8h*)&smB[row * 32 + 16 + half * 8];   // K = 16+half*8 .. +7
    }
#pragma unroll
    for (int i = 0; i < 2; i++)
#pragma unroll
      for (int j = 0; j < 4; j++)
        acc[i][j] = __builtin_amdgcn_wmma_f32_16x16x32_f16(
            false, af[i].v, false, bf[j].v, (short)0, acc[i][j], false, false);
    __syncthreads();
  }

  // epilogue + store. C/D layout: N = lane%16, M = v + 8*(lane/16)
#pragma unroll
  for (int j = 0; j < 4; j++) {
    int n = n0 + wn * 64 + j * 16 + r;
    if (n >= p.N) continue;
    float bia = p.bias ? p.bias[n] : 0.f;
#pragma unroll
    for (int i = 0; i < 2; i++) {
#pragma unroll
      for (int v = 0; v < 8; v++) {
        int m = m0 + wm * 32 + i * 16 + half * 8 + v;
        if (m >= p.M) continue;
        float c = acc[i][j][v] + bia;
        size_t co = row_off(m, p.ldc, p.cMode) + n;
        if (p.epi == 1) {
          c = (c + p.aux1[(size_t)m * p.N + n]) * SCALE_F;
        } else if (p.epi == 2) {
          c = (c + p.aux1[(size_t)m * p.N + n]) * SCALE_F;
          c = (c + C[co]) * SCALE_F;
        }
        C[co] = c;
      }
    }
  }
}

// ---------------------------------------------------------------------------
// Conv1d(K=3, HID -> 2*HID) + GLU fused.
// A: x_pad [B][102][HID] f32 (zero pad rows). W: pre-converted f16
// [l*3+tap][2048][1024]; the two 64x32 weight tiles per K-step are fetched by
// the Tensor Data Mover (wave 0 issues tensor_load_to_lds; TENSORcnt).
// Dual accumulators (a / g halves), writes yglu[m,o] = a*sigmoid(g).
// Block 256, tile 128(M)x64(N), K-loop 3*1024.
// ---------------------------------------------------------------------------
__global__ __launch_bounds__(256) void conv_glu_kernel(
    const float* __restrict__ xpad, const _Float16* __restrict__ w16,
    const float* __restrict__ convB, float* __restrict__ yglu, int layer) {
  __shared__ __align__(16) _Float16 smA [128 * 32];
  __shared__ __align__(16) _Float16 smBa[ 64 * 32];
  __shared__ __align__(16) _Float16 smBg[ 64 * 32];

  const int m0 = blockIdx.y * 128;
  const int n0 = blockIdx.x * 64;
  const int tid  = threadIdx.x;
  const int wave = tid >> 5;
  const int lane = tid & 31;
  const int wm = wave & 3;      // 4 waves over M (32 rows)
  const int wn = wave >> 2;     // 2 waves over N (32 cols)
  const int r    = lane & 15;
  const int half = lane >> 4;

  v8f accA[2][2] = {};
  v8f accG[2][2] = {};

  const unsigned ldsBa = lds_addr(&smBa[0]);
  const unsigned ldsBg = lds_addr(&smBg[0]);

  for (int kk = 0; kk < 3 * HID; kk += 32) {
    const int ks = kk >> 10;          // conv tap 0..2
    const int i0 = kk & 1023;         // input channel base
    // A tile: vectorized f32 -> f16 (always in range)
#pragma unroll
    for (int idx = tid; idx < 128 * 8; idx += 256) {
      int row = idx >> 3, c4 = (idx & 7) << 2;
      int mg = m0 + row;
      int b = mg / TRG, t = mg - b * TRG;
      float4 vv = *(const float4*)&xpad[(size_t)(b * TPAD + t + ks) * HID + i0 + c4];
      v4h h = { (_Float16)vv.x, (_Float16)vv.y, (_Float16)vv.z, (_Float16)vv.w };
      *(v4h*)&smA[row * 32 + c4] = h;
    }
    // B tiles via Tensor Data Mover: 64 rows x 32 cols f16, row stride 1024
    if (wave == 0) {
      const _Float16* srcA = w16 +
          ((((size_t)(layer * 3 + ks)) * 2048 + n0) << 10) + i0;
      tdm_load_tile_f16(ldsBa, srcA, 32, 64, 1024);
      tdm_load_tile_f16(ldsBg, srcA + ((size_t)1024 << 10), 32, 64, 1024);
      __builtin_amdgcn_s_wait_tensorcnt(0);
    }
    __syncthreads();

    F16Frag af[2], ba[2], bg[2];
    for (int i = 0; i < 2; i++) {
      int row = wm * 32 + i * 16 + r;
      af[i].h[0] = *(const v8h*)&smA[row * 32 + half * 8];
      af[i].h[1] = *(const v8h*)&smA[row * 32 + 16 + half * 8];
    }
    for (int j = 0; j < 2; j++) {
      int row = wn * 32 + j * 16 + r;
      ba[j].h[0] = *(const v8h*)&smBa[row * 32 + half * 8];
      ba[j].h[1] = *(const v8h*)&smBa[row * 32 + 16 + half * 8];
      bg[j].h[0] = *(const v8h*)&smBg[row * 32 + half * 8];
      bg[j].h[1] = *(const v8h*)&smBg[row * 32 + 16 + half * 8];
    }
#pragma unroll
    for (int i = 0; i < 2; i++)
#pragma unroll
      for (int j = 0; j < 2; j++) {
        accA[i][j] = __builtin_amdgcn_wmma_f32_16x16x32_f16(
            false, af[i].v, false, ba[j].v, (short)0, accA[i][j], false, false);
        accG[i][j] = __builtin_amdgcn_wmma_f32_16x16x32_f16(
            false, af[i].v, false, bg[j].v, (short)0, accG[i][j], false, false);
      }
    __syncthreads();
  }

  // GLU epilogue
#pragma unroll
  for (int j = 0; j < 2; j++) {
    int n = n0 + wn * 32 + j * 16 + r;
    float biasA = convB[layer * 2048 + n];
    float biasG = convB[layer * 2048 + 1024 + n];
#pragma unroll
    for (int i = 0; i < 2; i++) {
#pragma unroll
      for (int v = 0; v < 8; v++) {
        int m = m0 + wm * 32 + i * 16 + half * 8 + v;
        float a = accA[i][j][v] + biasA;
        float g = accG[i][j][v] + biasG;
        yglu[(size_t)m * HID + n] = a * (1.0f / (1.0f + __expf(-g)));
      }
    }
  }
}

// ---------------------------------------------------------------------------
// Host-side orchestration
// ---------------------------------------------------------------------------
static void launch_gemm(hipStream_t stream, dim3 grid, const GemmParams& p) {
  wmma_gemm_kernel<<<grid, 256, 0, stream>>>(p);
}

extern "C" void kernel_launch(void* const* d_in, const int* in_sizes, int n_in,
                              void* d_out, int out_size, void* d_ws, size_t ws_size,
                              hipStream_t stream) {
  const float* enc_conved   = (const float*)d_in[0];
  const float* enc_combined = (const float*)d_in[1];
  const float* tok_W  = (const float*)d_in[2];
  const float* tok_b  = (const float*)d_in[3];
  const float* pos_emb= (const float*)d_in[4];
  const float* tgt_W  = (const float*)d_in[5];
  const float* tgt_b  = (const float*)d_in[6];
  const float* e2h_W  = (const float*)d_in[7];
  const float* e2h_b  = (const float*)d_in[8];
  const float* h2e_W  = (const float*)d_in[9];
  const float* h2e_b  = (const float*)d_in[10];
  const float* ah2e_W = (const float*)d_in[11];
  const float* ah2e_b = (const float*)d_in[12];
  const float* ae2h_W = (const float*)d_in[13];
  const float* ae2h_b = (const float*)d_in[14];
  const float* fc_W   = (const float*)d_in[15];
  const float* fc_b   = (const float*)d_in[16];
  const float* conv_W = (const float*)d_in[17];
  const float* conv_b = (const float*)d_in[18];

  float* ws   = (float*)d_ws;
  float* mbuf = ws + OFF_M;
  float* ht   = ws + OFF_HT;
  float* tok  = ws + OFF_TOK;
  float* emb  = ws + OFF_EMB;     // embedded [B,TRG,EMB]
  float* xpad = ws + OFF_XPAD;    // conv_input, padded [B,102,HID]
  float* yglu = ws + OFF_YGLU;    // GLU output [B*TRG, HID]
  float* temb = ws + OFF_TEMB;    // [B*TRG, EMB] temp
  float* attn = ws + OFF_ATTN;    // [B, TRG, SRC]
  float* out  = (float*)d_out;

  _Float16* hb      = (_Float16*)(ws + HALF_BASE);
  _Float16* wconv16 = hb + H_WCONV;
  _Float16* e2h16   = hb + H_E2H;
  _Float16* h2e16   = hb + H_H2E;
  _Float16* ah2e16  = hb + H_AH2E;
  _Float16* ae2h16  = hb + H_AE2H;
  _Float16* fc16    = hb + H_FC;
  _Float16* encc16  = hb + H_ENCC;
  _Float16* enccb16 = hb + H_ENCCB;

  // --- per-call weight / encoder conversion to f16 (deterministic) ---
  cvt_convw_kernel<<<8192, 256, 0, stream>>>(conv_W, wconv16);
  cvt_f16_kernel<<<1024, 256, 0, stream>>>(e2h_W,  e2h16,  524288);
  cvt_f16_kernel<<<1024, 256, 0, stream>>>(h2e_W,  h2e16,  524288);
  cvt_f16_kernel<<<1024, 256, 0, stream>>>(ah2e_W, ah2e16, 524288);
  cvt_f16_kernel<<<1024, 256, 0, stream>>>(ae2h_W, ae2h16, 524288);
  cvt_f16_kernel<<<2048, 256, 0, stream>>>(fc_W,   fc16,   5120000);
  cvt_f16_kernel<<<2048, 256, 0, stream>>>(enc_conved,   encc16,  3276800);
  cvt_f16_kernel<<<2048, 256, 0, stream>>>(enc_combined, enccb16, 3276800);

  // 0) zero padded conv buffer (pad rows must be 0 every call)
  zero_fill_kernel<<<1024, 256, 0, stream>>>(xpad, (size_t)Bsz * TPAD * HID);

  // 1) m[b,s] = mean_e enc_conved[b,s,e]
  mean512_kernel<<<(Bsz * SRC) / 4, 128, 0, stream>>>(enc_conved, mbuf);

  // 2) hidden_target = m @ tgt_W^T + tgt_b     [64,100]
  small_gemm_kernel<<<(Bsz * TRG + 255) / 256, 256, 0, stream>>>(
      mbuf, tgt_W, tgt_b, ht, Bsz, TRG, SRC);

  // 3) tok = hidden_target @ tok_W^T + tok_b   [64,512]
  small_gemm_kernel<<<(Bsz * EMB + 255) / 256, 256, 0, stream>>>(
      ht, tok_W, tok_b, tok, Bsz, EMB, TRG);

  // 4) embedded = tok[:,None,:] + pos_emb
  embed_add_kernel<<<(Bsz * TRG * EMB + 255) / 256, 256, 0, stream>>>(tok, pos_emb, emb);

  // 5) conv_input = e2h(embedded) into padded buffer
  {
    GemmParams p = { emb, 0, EMB, 0,
                     e2h16, 0, (size_t)EMB, 1,
                     e2h_b,
                     xpad, 0, HID, 1,
                     nullptr, Bsz * TRG, HID, EMB, 0 };
    launch_gemm(stream, dim3(HID / 128, (Bsz * TRG) / 128, 1), p);
  }

  // 6) decoder layers
  for (int l = 0; l < NLAY; l++) {
    conv_glu_kernel<<<dim3(HID / 64, (Bsz * TRG) / 128), 256, 0, stream>>>(
        xpad, wconv16, conv_b, yglu, l);

    // combined = (ah2e(yglu) + embedded) * SCALE -> temb
    {
      GemmParams p = { yglu, 0, HID, 0,
                       ah2e16, 0, (size_t)HID, 1,
                       ah2e_b,
                       temb, 0, EMB, 0,
                       emb, Bsz * TRG, EMB, HID, 1 };
      launch_gemm(stream, dim3(EMB / 128, (Bsz * TRG) / 128, 1), p);
    }

    // energy[b,t,s] = combined[b,t,:] . enc_conved[b,s,:]   (batched)
    {
      GemmParams p = { temb, (size_t)TRG * EMB, EMB, 0,
                       encc16, (size_t)SRC * EMB, (size_t)EMB, 1,
                       nullptr,
                       attn, (size_t)TRG * SRC, SRC, 0,
                       nullptr, TRG, SRC, EMB, 0 };
      launch_gemm(stream, dim3(1, 1, Bsz), p);
    }

    softmax100_kernel<<<(Bsz * TRG) / 4, 128, 0, stream>>>(attn);

    // attended[b,t,e] = attn[b,t,:] @ enc_combined[b,:,e]   (untransposed W)
    {
      GemmParams p = { attn, (size_t)TRG * SRC, SRC, 0,
                       enccb16, (size_t)SRC * EMB, 1, (size_t)EMB,
                       nullptr,
                       temb, (size_t)TRG * EMB, EMB, 0,
                       nullptr, TRG, EMB, SRC, 0 };
      launch_gemm(stream, dim3(EMB / 128, 1, Bsz), p);
    }

    // x_pad = ((ae2h(attended) + yglu)*SCALE + x_pad)*SCALE
    {
      GemmParams p = { temb, 0, EMB, 0,
                       ae2h16, 0, (size_t)EMB, 1,
                       ae2h_b,
                       xpad, 0, HID, 1,
                       yglu, Bsz * TRG, HID, EMB, 2 };
      launch_gemm(stream, dim3(HID / 128, (Bsz * TRG) / 128, 1), p);
    }
  }

  // 7) conved = h2e(conv_input) -> temb
  {
    GemmParams p = { xpad, 0, HID, 1,
                     h2e16, 0, (size_t)HID, 1,
                     h2e_b,
                     temb, 0, EMB, 0,
                     nullptr, Bsz * TRG, EMB, HID, 0 };
    launch_gemm(stream, dim3(EMB / 128, (Bsz * TRG) / 128, 1), p);
  }

  // 8) output = fc(conved)
  {
    GemmParams p = { temb, 0, EMB, 0,
                     fc16, 0, (size_t)EMB, 1,
                     fc_b,
                     out, 0, OUTN, 0,
                     nullptr, Bsz * TRG, OUTN, EMB, 0 };
    launch_gemm(stream, dim3((OUTN + 127) / 128, (Bsz * TRG) / 128, 1), p);
  }

  // 9) append last-layer attention to d_out
  (void)hipMemcpyAsync(out + (size_t)Bsz * TRG * OUTN, attn,
                       (size_t)Bsz * TRG * SRC * sizeof(float),
                       hipMemcpyDeviceToDevice, stream);
}
